// Self_Attn_63556926046985
// MI455X (gfx1250) — matrix operand order repaired and checked by reading, not compiled, not run on hardware
//
#include <hip/hip_runtime.h>
#include <hip/hip_bf16.h>

// ---------------------------------------------------------------------------
// SAGAN self-attention, B=2 C=64 N=96*96=9216, softmax over dim=1 (queries).
// Fused (never materialize the 680MB energy/attn matrix):
//   K1: q[B][8][N], k[B][8][N] (f32), V[B][64][N] (f16) projections
//   K2: per-column stats M[n]=max_n' E[n',n], Dinv[n]=1/sum exp(E[n',n]-M[n])
//   K3: per 16-wide m-tile: E row-tile via v_wmma_f32_16x16x4_f32 (K=8),
//       P=exp(E-M)*Dinv -> f16 via LDS re-swizzle, out tile accumulated with
//       v_wmma_f32_16x16x32_f16 against V^T; epilogue gamma*out + x.
// ---------------------------------------------------------------------------

typedef __attribute__((ext_vector_type(16))) _Float16 v16h;
typedef __attribute__((ext_vector_type(8)))  _Float16 v8h;
typedef __attribute__((ext_vector_type(8)))  float    v8f;
typedef __attribute__((ext_vector_type(4)))  float    v4f;
typedef __attribute__((ext_vector_type(2)))  float    v2f;

#define NB   2
#define CC   64
#define NN   9216   // 96*96

// ---------------- Kernel 1: q/k/v projections (1x1 convs) -------------------
// grid (36, 80, 2), block 256. rows 0..7 -> q, 8..15 -> k, 16..79 -> v(f16)
__global__ __launch_bounds__(256) void qkv_proj(
    const float* __restrict__ x,
    const float* __restrict__ Wq, const float* __restrict__ bq,
    const float* __restrict__ Wk, const float* __restrict__ bk,
    const float* __restrict__ Wv, const float* __restrict__ bv,
    float* __restrict__ qT, float* __restrict__ kT, _Float16* __restrict__ Vh)
{
    const int n   = blockIdx.x * 256 + threadIdx.x;
    const int row = blockIdx.y;            // 0..79
    const int b   = blockIdx.z;
    const float* xb = x + (size_t)b * CC * NN;

    const float* w;
    float acc;
    if (row < 8)       { w = Wq + row * CC;        acc = bq[row];      }
    else if (row < 16) { w = Wk + (row - 8) * CC;  acc = bk[row - 8];  }
    else               { w = Wv + (row - 16) * CC; acc = bv[row - 16]; }

    #pragma unroll 8
    for (int c = 0; c < CC; ++c)
        acc = fmaf(w[c], xb[(size_t)c * NN + n], acc);   // coalesced in n

    if (row < 8)       qT[((size_t)b * 8 + row) * NN + n]        = acc;
    else if (row < 16) kT[((size_t)b * 8 + (row - 8)) * NN + n]  = acc;
    else               Vh[((size_t)b * CC + (row - 16)) * NN + n] = (_Float16)acc;
}

// ---------------- Kernel 2: column softmax stats ---------------------------
// softmax over dim=1 => for each column n: reduce over rows n' of E[n',n].
// grid (36, 2), block 256: thread owns one column, q rows staged in LDS.
#define CH 512
__global__ __launch_bounds__(256) void col_stats(
    const float* __restrict__ qT, const float* __restrict__ kT,
    float* __restrict__ Mx, float* __restrict__ Dinv)
{
    __shared__ float qs[CH * 8];
    const int tid  = threadIdx.x;
    const int b    = blockIdx.y;
    const int mcol = blockIdx.x * 256 + tid;
    const float* q = qT + (size_t)b * 8 * NN;
    const float* k = kT + (size_t)b * 8 * NN;

    float kr[8];
    #pragma unroll
    for (int o = 0; o < 8; ++o) kr[o] = k[(size_t)o * NN + mcol];

    float m = -INFINITY, d = 0.0f;
    for (int n0 = 0; n0 < NN; n0 += CH) {
        __syncthreads();
        for (int idx = tid; idx < CH * 8; idx += 256) {
            const int o = idx / CH, i = idx % CH;         // coalesced in i
            qs[i * 8 + o] = q[(size_t)o * NN + n0 + i];
        }
        __syncthreads();
        #pragma unroll 4
        for (int i = 0; i < CH; ++i) {
            v4f q0 = *(const v4f*)&qs[i * 8];             // b128 LDS broadcast
            v4f q1 = *(const v4f*)&qs[i * 8 + 4];
            float e = q0.x*kr[0] + q0.y*kr[1] + q0.z*kr[2] + q0.w*kr[3]
                    + q1.x*kr[4] + q1.y*kr[5] + q1.z*kr[6] + q1.w*kr[7];
            if (e > m) { d *= __expf(m - e); m = e; }     // online softmax
            d += __expf(e - m);
        }
    }
    Mx[(size_t)b * NN + mcol]   = m;
    Dinv[(size_t)b * NN + mcol] = 1.0f / d;
}

// ---------------- Kernel 3: fused attention * V + residual -----------------
// grid (N/16, 2), block 32 (one wave; WMMA needs EXEC all-ones, no divergence)
__global__ __launch_bounds__(32) void attn_out(
    const float* __restrict__ qT, const float* __restrict__ kT,
    const _Float16* __restrict__ Vh,
    const float* __restrict__ Mx, const float* __restrict__ Dinv,
    const float* __restrict__ x, const float* __restrict__ gamma,
    float* __restrict__ out)
{
    __shared__ _Float16 Pt[16 * 32];      // P tile, row-major [m][k], 1KB
    const int lane = threadIdx.x;
    const int hi   = lane >> 4;           // lane>=16 half
    const int ml   = lane & 15;
    const int b    = blockIdx.y;
    const int m0   = blockIdx.x * 16;

    const float*    q  = qT + (size_t)b * 8 * NN;
    const float*    k  = kT + (size_t)b * 8 * NN;
    const _Float16* v  = Vh + (size_t)b * CC * NN;
    const float*    mx = Mx + (size_t)b * NN;
    const float*    dv = Dinv + (size_t)b * NN;

    // A operand (f32 16x4): lane<16 holds K=0,1 ; lane>=16 K=2,3 per VGPR pair
    v2f a_lo, a_hi;
    a_lo.x = q[(size_t)(0 + 2 * hi) * NN + m0 + ml];
    a_lo.y = q[(size_t)(1 + 2 * hi) * NN + m0 + ml];
    a_hi.x = q[(size_t)(4 + 2 * hi) * NN + m0 + ml];
    a_hi.y = q[(size_t)(5 + 2 * hi) * NN + m0 + ml];

    v8f acc[4] = { v8f{}, v8f{}, v8f{}, v8f{} };   // 4 c-tiles of D (m x c)

    for (int n0 = 0; n0 < NN; n0 += 32) {
        // ---- E row-tile [16m x 32n] via f32 WMMA, P -> LDS ----
        #pragma unroll
        for (int nt = 0; nt < 2; ++nt) {
            const int n = n0 + nt * 16 + ml;
            v2f b_lo, b_hi;                         // B (f32 4x16) layout
            b_lo.x = k[(size_t)(0 + 2 * hi) * NN + n];
            b_lo.y = k[(size_t)(1 + 2 * hi) * NN + n];
            b_hi.x = k[(size_t)(4 + 2 * hi) * NN + n];
            b_hi.y = k[(size_t)(5 + 2 * hi) * NN + n];
            v8f e = {};
            e = __builtin_amdgcn_wmma_f32_16x16x4_f32(false, a_lo, false, b_lo,
                                                      (short)0, e, false, false);
            e = __builtin_amdgcn_wmma_f32_16x16x4_f32(false, a_hi, false, b_hi,
                                                      (short)0, e, false, false);
            const float cm = mx[n];                 // per-column stats
            const float cd = dv[n];
            #pragma unroll
            for (int r = 0; r < 8; ++r) {           // D layout: row m = r + 8*hi
                const float p = __expf(e[r] - cm) * cd;
                Pt[(r + 8 * hi) * 32 + nt * 16 + ml] = (_Float16)p;
            }
        }
        __syncthreads();

        // ---- A operand of P (f16 16x32): lane ml row; K = e + 16*hi consecutive
        v8h plo = *(const v8h*)&Pt[ml * 32 + 16 * hi];
        v8h phi = *(const v8h*)&Pt[ml * 32 + 16 * hi + 8];
        v16h pa = __builtin_shufflevector(plo, phi,
                    0,1,2,3,4,5,6,7,8,9,10,11,12,13,14,15);

        // ---- B operand = V^T chunk (f16 32x16): K = e + 16*hi consecutive
        #pragma unroll
        for (int ct = 0; ct < 4; ++ct) {
            const _Float16* vp = v + (size_t)(ct * 16 + ml) * NN + n0 + 16 * hi;
            v8h b0 = *(const v8h*)vp;
            v8h b1 = *(const v8h*)(vp + 8);
            v16h vb = __builtin_shufflevector(b0, b1,
                        0,1,2,3,4,5,6,7,8,9,10,11,12,13,14,15);
            acc[ct] = __builtin_amdgcn_wmma_f32_16x16x32_f16(
                          false, pa, false, vb, (short)0, acc[ct], false, false);
        }
        __syncthreads();
    }

    // ---- epilogue: out[b][c][m] = gamma*acc + x ; rows of D contiguous in m
    const float g = gamma[0];
    #pragma unroll
    for (int ct = 0; ct < 4; ++ct) {
        const int c = ct * 16 + ml;
        const size_t base = ((size_t)b * CC + c) * NN + m0 + 8 * hi;
        v4f x0 = *(const v4f*)&x[base];
        v4f x1 = *(const v4f*)&x[base + 4];
        v4f o0, o1;
        #pragma unroll
        for (int r = 0; r < 4; ++r) {
            o0[r] = g * acc[ct][r]     + x0[r];
            o1[r] = g * acc[ct][r + 4] + x1[r];
        }
        *(v4f*)&out[base]     = o0;
        *(v4f*)&out[base + 4] = o1;
    }
}

// ---------------------------------------------------------------------------
extern "C" void kernel_launch(void* const* d_in, const int* in_sizes, int n_in,
                              void* d_out, int out_size, void* d_ws, size_t ws_size,
                              hipStream_t stream) {
    const float* x     = (const float*)d_in[0];
    const float* Wq    = (const float*)d_in[1];
    const float* bq    = (const float*)d_in[2];
    const float* Wk    = (const float*)d_in[3];
    const float* bk    = (const float*)d_in[4];
    const float* Wv    = (const float*)d_in[5];
    const float* bv    = (const float*)d_in[6];
    const float* gamma = (const float*)d_in[7];
    float* out = (float*)d_out;

    // workspace layout (~3.6 MB total)
    float*    qT   = (float*)d_ws;                  // [2][8][N] f32
    float*    kT   = qT + (size_t)NB * 8 * NN;      // [2][8][N] f32
    _Float16* Vh   = (_Float16*)(kT + (size_t)NB * 8 * NN); // [2][64][N] f16
    float*    Mx   = (float*)(Vh + (size_t)NB * CC * NN);   // [2][N]
    float*    Dinv = Mx + (size_t)NB * NN;                  // [2][N]

    qkv_proj<<<dim3(NN / 256, 80, NB), 256, 0, stream>>>(
        x, Wq, bq, Wk, bk, Wv, bv, qT, kT, Vh);

    col_stats<<<dim3(NN / 256, NB), 256, 0, stream>>>(qT, kT, Mx, Dinv);

    attn_out<<<dim3(NN / 16, NB), 32, 0, stream>>>(
        qT, kT, Vh, Mx, Dinv, x, gamma, out);
}
